// LSTM_49211735278135
// MI455X (gfx1250) — compile-verified, matrix operand order checked by reading, not run on hardware
//
#include <hip/hip_runtime.h>
#include <hip/hip_bf16.h>

// -----------------------------------------------------------------------------
// 3-layer LSTM (B=1024, T=512, H=128) + FC(128->1) for MI455X (gfx1250).
//
// Persistent workgroups, one per 16-row batch tile (64 WGs), no cross-WG sync.
// Per step: gates[16x512] = [x_t | h_{t-1}](16x256 bf16) @ W(256x512 bf16)
// via v_wmma_f32_16x16x32_bf16 (32 per wave per step), fp32 accum/state.
// Weights are staged once per layer into LDS, then ALL B fragments for this
// wave's gate columns are hoisted into persistent VGPRs (4 gates x 8 ksteps
// x 8 VGPRs = 256 VGPRs; gfx1250's >256-VGPR file + s_set_vgpr_msb makes this
// viable, confirmed spill-free by the round-2 asm). Steady-state LDS traffic
// is then only the 16 A-fragment ds_load_b128 per step + h stores.
// Bias enters as the C operand of the ks=0 WMMA (no accumulator-init movs).
// Wave w owns gate columns {j, j+128, j+256, j+384}, j in [16w,16w+16): the
// i/f/g/o pre-activations for one (m,j) land in the same lane/component of 4
// accumulators -> register-resident cell update, c = one v8f per lane.
// Layer-to-layer h sequences stream through HBM as bf16; per-step input tiles
// are staged with GLOBAL_LOAD_ASYNC_TO_LDS_B128 (ASYNCcnt) to overlap WMMA.
// -----------------------------------------------------------------------------

#define B_SZ   1024
#define T_SZ   512
#define DIN0   6
#define H_SZ   128
#define NGATE  512            // 4*H
#define KTOT   256            // [input(128, zero-padded for layer0) | hidden(128)]
#define MT     16             // batch rows per workgroup
#define RS     264            // LDS row stride in bf16 (256 + 8 pad -> 528B rows)
#define NBLK   (B_SZ / MT)    // 64 workgroups
#define NTHR   256            // 8 waves (wave32)

typedef __bf16 bf16_t;
typedef __attribute__((ext_vector_type(16))) __bf16 v16bf;
typedef __attribute__((ext_vector_type(8)))  __bf16 bf16x8;
typedef __attribute__((ext_vector_type(8)))  float  v8f;

union ABfrag { v16bf v; bf16x8 h[2]; };

// LDS layout (byte offsets inside dynamic shared segment)
#define W_OFF   0u
#define A0_OFF  ((unsigned)(NGATE * RS * 2))
#define A1_OFF  (A0_OFF + (unsigned)(MT * RS * 2))

__device__ __forceinline__ float fastrcp_(float x) {
    return __builtin_amdgcn_rcpf(x);           // v_rcp_f32
}
__device__ __forceinline__ float sigmoidf_(float x) {
    return fastrcp_(1.0f + __expf(-x));
}
__device__ __forceinline__ float tanhf_(float x) {
    // tanh(x) = 1 - 2/(exp(2x)+1); saturates correctly at +-1
    return 1.0f - 2.0f * fastrcp_(__expf(2.0f * x) + 1.0f);
}

// Layers 1,2: async-copy the 16x128 bf16 input tile for timestep t straight
// into the A buffer (ASYNCcnt-tracked, no VGPR transit). 256 threads x 16B.
__device__ __forceinline__ void stage_input_async(unsigned abuf_off,
                                                  const bf16_t* __restrict__ inseq,
                                                  int b0, int t, int tid) {
    int m = tid >> 4, ch = tid & 15;
    unsigned lds_off = abuf_off + (unsigned)(m * RS + ch * 8) * 2u;
    unsigned goff = (unsigned)(((((size_t)(b0 + m) * T_SZ + t) * H_SZ) + ch * 8) * 2);
    asm volatile("global_load_async_to_lds_b128 %0, %1, %2"
                 :: "v"(lds_off), "v"(goff), "s"(inseq)
                 : "memory");
}

// Layer 0: x is [B,T,6] f32; needs f32->bf16 conversion, so plain path.
__device__ __forceinline__ void stage_input_l0(bf16_t* __restrict__ abuf,
                                               const float* __restrict__ x,
                                               int b0, int t, int tid) {
    if (tid < MT * DIN0) {
        int m = tid / DIN0, k = tid % DIN0;
        abuf[m * RS + k] = (bf16_t)x[((size_t)(b0 + m) * T_SZ + t) * DIN0 + k];
    }
}

__global__ __launch_bounds__(NTHR, 1)
void lstm_wmma_kernel(const float* __restrict__ x,
                      const float* __restrict__ wih0, const float* __restrict__ whh0,
                      const float* __restrict__ bih0, const float* __restrict__ bhh0,
                      const float* __restrict__ wih1, const float* __restrict__ whh1,
                      const float* __restrict__ bih1, const float* __restrict__ bhh1,
                      const float* __restrict__ wih2, const float* __restrict__ whh2,
                      const float* __restrict__ bih2, const float* __restrict__ bhh2,
                      const float* __restrict__ fcw,  const float* __restrict__ fcb,
                      float* __restrict__ out,
                      bf16_t* __restrict__ seq0, bf16_t* __restrict__ seq1)
{
    extern __shared__ __align__(16) char smem_raw[];
    bf16_t* Wl = (bf16_t*)smem_raw;          // [NGATE][RS]  weights, W[n][k]
    bf16_t* A0 = Wl + NGATE * RS;            // [MT][RS]     A tile, double-buffered
    bf16_t* A1 = A0 + MT * RS;

    const int tid  = threadIdx.x;
    const int lane = tid & 31;
    const int wv   = tid >> 5;               // 0..7, owns hidden j in [16wv,16wv+16)
    const int ln   = lane & 15;
    const int kh   = lane >> 4;              // khalf selector (A/B frag layout)
    const int b0   = blockIdx.x * MT;

    const float* WIH[3] = {wih0, wih1, wih2};
    const float* WHH[3] = {whh0, whh1, whh2};
    const float* BIH[3] = {bih0, bih1, bih2};
    const float* BHH[3] = {bhh0, bhh1, bhh2};

    for (int layer = 0; layer < 3; ++layer) {
        const int Din = (layer == 0) ? DIN0 : H_SZ;
        const float*  wih = WIH[layer];
        const float*  whh = WHH[layer];
        const bf16_t* inseq  = (layer == 0) ? nullptr : ((layer == 1) ? seq0 : seq1);
        bf16_t*       outseq = (layer == 0) ? seq0    : ((layer == 1) ? seq1 : nullptr);

        // ---- stage weights into LDS: W[n][k] = (k<128 ? w_ih[n][k] : w_hh[n][k-128])
        for (int idx = tid; idx < NGATE * KTOT; idx += NTHR) {
            int n = idx >> 8, k = idx & 255;
            float v;
            if (k < H_SZ) v = (k < Din) ? wih[n * Din + k] : 0.0f;
            else          v = whh[n * H_SZ + (k - H_SZ)];
            Wl[n * RS + k] = (bf16_t)v;
        }
        // ---- zero both A buffers (h_{-1}=0, layer-0 pad columns stay 0)
        for (int idx = tid; idx < 2 * MT * RS; idx += NTHR)
            A0[idx] = (bf16_t)0.0f;
        __syncthreads();

        // ---- hoist ALL B fragments into persistent VGPRs (4g x 8ks x 8 = 256
        //      VGPRs): loop-invariant across all 512 timesteps. The gfx1250
        //      allocator handles >256 VGPRs via s_set_vgpr_msb (no spills).
        ABfrag bfr[4][8];
        #pragma unroll
        for (int g = 0; g < 4; ++g)
            #pragma unroll
            for (int ks = 0; ks < 8; ++ks) {
                const bf16_t* bp = Wl + (g * H_SZ + wv * 16 + ln) * RS
                                      + ks * 32 + kh * 8;
                bfr[g][ks].h[0] = *(const bf16x8*)(bp);
                bfr[g][ks].h[1] = *(const bf16x8*)(bp + 16);
            }

        // ---- stage input for t=0 into A0
        if (layer == 0) stage_input_l0(A0, x, b0, 0, tid);
        else            stage_input_async(A0_OFF, inseq, b0, 0, tid);

        // ---- per-lane bias vectors: C operand of the ks=0 WMMA each step
        v8f biasv[4];
        #pragma unroll
        for (int g = 0; g < 4; ++g) {
            int n = g * H_SZ + wv * 16 + ln;
            float bsum = BIH[layer][n] + BHH[layer][n];
            #pragma unroll
            for (int r = 0; r < 8; ++r) biasv[g][r] = bsum;
        }
        // ---- c state: lane holds c[m][j], m = kh*8 + r, j = wv*16 + ln
        v8f creg;
        #pragma unroll
        for (int r = 0; r < 8; ++r) creg[r] = 0.0f;

        asm volatile("s_wait_asynccnt 0x0" ::: "memory");
        __syncthreads();

        bf16_t* cur = A0; unsigned cur_off = A0_OFF;
        bf16_t* nxt = A1; unsigned nxt_off = A1_OFF;

        for (int t = 0; t < T_SZ; ++t) {
            v8f acc[4];
            // ---- gates = bias + A(16x256) @ W(256x512-slice): 32 WMMAs/wave,
            //      B entirely register-resident, only A comes from LDS.
            #pragma unroll
            for (int ks = 0; ks < 8; ++ks) {
                ABfrag a;
                const bf16_t* ap = cur + ln * RS + ks * 32 + kh * 8;
                a.h[0] = *(const bf16x8*)(ap);        // K = ks*32 + kh*8 + 0..7
                a.h[1] = *(const bf16x8*)(ap + 16);   // K = ks*32 + 16 + kh*8 + 0..7
                #pragma unroll
                for (int g = 0; g < 4; ++g) {
                    v8f cin = (ks == 0) ? biasv[g] : acc[g];
                    acc[g] = __builtin_amdgcn_wmma_f32_16x16x32_bf16(
                        false, a.v, false, bfr[g][ks].v, (short)0, cin, false, false);
                }
            }

            // ---- register-resident cell update; i/f/g/o share lane+component
            #pragma unroll
            for (int r = 0; r < 8; ++r) {
                float iv = sigmoidf_(acc[0][r]);
                float fv = sigmoidf_(acc[1][r]);
                float gv = tanhf_   (acc[2][r]);
                float ov = sigmoidf_(acc[3][r]);
                float cv = fv * creg[r] + iv * gv;
                creg[r] = cv;
                float hv = ov * tanhf_(cv);
                bf16_t hb = (bf16_t)hv;
                int m = kh * 8 + r;
                int j = wv * 16 + ln;
                nxt[m * RS + H_SZ + j] = hb;                       // h -> next A tile
                if (outseq)                                        // h -> next layer seq
                    outseq[((size_t)(b0 + m) * T_SZ + t) * H_SZ + j] = hb;
            }

            // ---- stage x_{t+1} into the other half of nxt (async for L1/L2)
            if (t + 1 < T_SZ) {
                if (layer == 0) stage_input_l0(nxt, x, b0, t + 1, tid);
                else            stage_input_async(nxt_off, inseq, b0, t + 1, tid);
            }

            bf16_t* tp = cur; cur = nxt; nxt = tp;
            unsigned to = cur_off; cur_off = nxt_off; nxt_off = to;

            asm volatile("s_wait_asynccnt 0x0" ::: "memory");
            __syncthreads();
        }

        // ---- final FC on h_{T-1} (sits in cur's hidden half), layer 2 only
        if (layer == 2) {
            if (tid < MT) {
                int m = tid;
                float s = fcb[0];
                for (int j = 0; j < H_SZ; ++j)
                    s += (float)cur[m * RS + H_SZ + j] * fcw[j];
                out[b0 + m] = s;
            }
        }
        __syncthreads();
    }
}

extern "C" void kernel_launch(void* const* d_in, const int* in_sizes, int n_in,
                              void* d_out, int out_size, void* d_ws, size_t ws_size,
                              hipStream_t stream) {
    const float* xx   = (const float*)d_in[0];
    const float* wih0 = (const float*)d_in[1];
    const float* whh0 = (const float*)d_in[2];
    const float* bih0 = (const float*)d_in[3];
    const float* bhh0 = (const float*)d_in[4];
    const float* wih1 = (const float*)d_in[5];
    const float* whh1 = (const float*)d_in[6];
    const float* bih1 = (const float*)d_in[7];
    const float* bhh1 = (const float*)d_in[8];
    const float* wih2 = (const float*)d_in[9];
    const float* whh2 = (const float*)d_in[10];
    const float* bih2 = (const float*)d_in[11];
    const float* bhh2 = (const float*)d_in[12];
    const float* fcw  = (const float*)d_in[13];
    const float* fcb  = (const float*)d_in[14];
    float* out = (float*)d_out;

    // workspace: two bf16 h-sequences [B, T, H] (128MB each)
    bf16_t* seq0 = (bf16_t*)d_ws;
    bf16_t* seq1 = seq0 + (size_t)B_SZ * T_SZ * H_SZ;

    const int smem = (NGATE * RS + 2 * MT * RS) * (int)sizeof(bf16_t); // 287,232 B
    hipFuncSetAttribute((const void*)lstm_wmma_kernel,
                        hipFuncAttributeMaxDynamicSharedMemorySize, smem);

    lstm_wmma_kernel<<<dim3(NBLK), dim3(NTHR), smem, stream>>>(
        xx, wih0, whh0, bih0, bhh0, wih1, whh1, bih1, bhh1,
        wih2, whh2, bih2, bhh2, fcw, fcb, out, seq0, seq1);
}
